// ProGenBase_19439021981701
// MI455X (gfx1250) — matrix operand order repaired and checked by reading, not compiled, not run on hardware
//
#include <hip/hip_runtime.h>
#include <hip/hip_bf16.h>
#include <math.h>

typedef __attribute__((ext_vector_type(16))) __bf16 v16bf;
typedef __attribute__((ext_vector_type(8)))  __bf16 v8bf;
typedef __attribute__((ext_vector_type(4)))  __bf16 v4bf;
typedef __attribute__((ext_vector_type(8)))  float  v8f;

// ---------------- model constants ----------------
constexpr int SEQn   = 2048;
constexpr int DIMn   = 1024;
constexpr int INNERn = 1024;
constexpr int HEADSn = 16;
constexpr int DHn    = 64;
constexpr int WSn    = 256;   // local attention window
constexpr int KWn    = 512;   // lookback keys per window (2*WS)
constexpr int QBn    = 32;    // q rows per attention block
constexpr float MASKV = -1e10f;

// ---------------- WMMA helpers ----------------
__device__ __forceinline__ v8f wmma_bf16(v16bf a, v16bf b, v8f c) {
  return __builtin_amdgcn_wmma_f32_16x16x32_bf16(false, a, false, b, (short)0, c, false, false);
}

__device__ __forceinline__ v16bf pack16(v8bf lo, v8bf hi) {
  v16bf r;
#pragma unroll
  for (int i = 0; i < 8; ++i) { r[i] = lo[i]; r[i + 8] = hi[i]; }
  return r;
}

__device__ __forceinline__ v8bf cvt8(float4 a, float4 b) {
  v8bf r;
  r[0] = (__bf16)a.x; r[1] = (__bf16)a.y; r[2] = (__bf16)a.z; r[3] = (__bf16)a.w;
  r[4] = (__bf16)b.x; r[5] = (__bf16)b.y; r[6] = (__bf16)b.z; r[7] = (__bf16)b.w;
  return r;
}

__device__ __forceinline__ v16bf cvt16f(const float* p0, const float* p1) {
  // 8 floats at p0 -> elems 0..7, 8 floats at p1 -> elems 8..15 (fragment halves)
  float4 a0 = *(const float4*)p0, a1 = *(const float4*)(p0 + 4);
  float4 b0 = *(const float4*)p1, b1 = *(const float4*)(p1 + 4);
  return pack16(cvt8(a0, a1), cvt8(b0, b1));
}

// ---------------- generic tiled GEMM ----------------
// C(MxN) = A(MxK) @ B(KxN) with compile-time fused epilogue.
constexpr int BMt = 128, BNt = 128, BKt = 32, LDSS = 48; // bf16 row stride 96B (16B aligned)
constexpr int GF_TRIL = 1, GF_GELU = 2, GF_BIAS = 4, GF_RESID = 8;

template<int FLAGS>
__global__ __launch_bounds__(256) void gemm_bf16_kernel(
    const float* __restrict__ A, const float* __restrict__ B,
    const float* __restrict__ bias, const float* __restrict__ resid,
    float* __restrict__ C, int M, int N, int K)
{
  constexpr bool TRIL  = (FLAGS & GF_TRIL)  != 0;
  constexpr bool GELU  = (FLAGS & GF_GELU)  != 0;
  constexpr bool BIAS  = (FLAGS & GF_BIAS)  != 0;
  constexpr bool RESID = (FLAGS & GF_RESID) != 0;

  __shared__ __align__(16) __bf16 As[2][BMt * LDSS];
  __shared__ __align__(16) __bf16 Bs[2][BNt * LDSS];

  const int tid  = threadIdx.x;
  const int wave = tid >> 5, lane = tid & 31;
  const int l16  = lane & 15;
  const bool hi  = lane >= 16;
  const int wm   = wave >> 2;       // 2 wave-rows x 4 wave-cols
  const int wn   = wave & 3;
  const int m0   = blockIdx.y * BMt;
  const int n0   = blockIdx.x * BNt;

  // A staging: thread owns A[m0+a_m][k0+a_k .. +15]   (fully coalesced b128 loads)
  const int a_m = tid >> 1;
  const int a_k = (tid & 1) * 16;
  // B staging: thread owns B[k0+b_k .. +3][n0+b_n .. +3] (coalesced b128 loads, b64 transp. stores)
  const int b_k = (tid >> 5) * 4;
  const int b_n = (tid & 31) * 4;

  float4 ra[4], rb[4];

  auto loadA = [&](int k0) {
    const float* p = &A[(size_t)(m0 + a_m) * K + k0 + a_k];
#pragma unroll
    for (int r = 0; r < 4; ++r) ra[r] = *(const float4*)(p + 4 * r);
    if (TRIL) {
      int mrow = m0 + a_m;
#pragma unroll
      for (int r = 0; r < 4; ++r) {
        float* f = (float*)&ra[r];
#pragma unroll
        for (int c = 0; c < 4; ++c)
          if (k0 + a_k + 4 * r + c > mrow) f[c] = 0.f;
      }
    }
  };
  auto loadB = [&](int k0) {
#pragma unroll
    for (int r = 0; r < 4; ++r)
      rb[r] = *(const float4*)&B[(size_t)(k0 + b_k + r) * N + n0 + b_n];
  };
  auto storeTiles = [&](int buf) {
    *(v8bf*)&As[buf][a_m * LDSS + a_k]     = cvt8(ra[0], ra[1]);
    *(v8bf*)&As[buf][a_m * LDSS + a_k + 8] = cvt8(ra[2], ra[3]);
#pragma unroll
    for (int c = 0; c < 4; ++c) {
      v4bf pb;
#pragma unroll
      for (int r = 0; r < 4; ++r) pb[r] = (__bf16)((const float*)&rb[r])[c];
      *(v4bf*)&Bs[buf][(b_n + c) * LDSS + b_k] = pb;
    }
  };

  v8f acc[4][2];
  const v8f vz = {};
#pragma unroll
  for (int i = 0; i < 4; ++i)
#pragma unroll
    for (int j = 0; j < 2; ++j) acc[i][j] = vz;

  const int nk = K / BKt;
  loadA(0); loadB(0);
  for (int kt = 0; kt < nk; ++kt) {
    const int buf = kt & 1;
    storeTiles(buf);
    __syncthreads();
    if (kt + 1 < nk) { loadA((kt + 1) * BKt); loadB((kt + 1) * BKt); }

    v16bf af[4], bfv[2];
#pragma unroll
    for (int am = 0; am < 4; ++am) {
      const __bf16* p = &As[buf][(wm * 64 + am * 16 + l16) * LDSS + (hi ? 8 : 0)];
      af[am] = pack16(*(const v8bf*)p, *(const v8bf*)(p + 16));
    }
#pragma unroll
    for (int bn = 0; bn < 2; ++bn) {
      const __bf16* p = &Bs[buf][(wn * 32 + bn * 16 + l16) * LDSS + (hi ? 8 : 0)];
      bfv[bn] = pack16(*(const v8bf*)p, *(const v8bf*)(p + 16));
    }
#pragma unroll
    for (int am = 0; am < 4; ++am)
#pragma unroll
      for (int bn = 0; bn < 2; ++bn)
        acc[am][bn] = wmma_bf16(af[am], bfv[bn], acc[am][bn]);
    // single barrier per tile: double buffer makes the trailing sync unnecessary
  }

  // epilogue: elem r -> M = base + (hi?8:0) + r, N = base + l16
#pragma unroll
  for (int am = 0; am < 4; ++am)
#pragma unroll
    for (int bn = 0; bn < 2; ++bn)
#pragma unroll
      for (int r = 0; r < 8; ++r) {
        int m = m0 + wm * 64 + am * 16 + (hi ? 8 : 0) + r;
        int n = n0 + wn * 32 + bn * 16 + l16;
        float v = acc[am][bn][r];
        if (BIAS)  v += bias[n];
        if (GELU)  v = 0.5f * v * (1.f + erff(v * 0.70710678118654752f));
        if (RESID) v += resid[(size_t)m * N + n];
        C[(size_t)m * N + n] = v;
      }
}

// ---------------- local windowed attention (WMMA for QK^T and PV) ----------------
__global__ __launch_bounds__(256) void attn_kernel(
    const float* __restrict__ qkv, float* __restrict__ out)
{
  extern __shared__ char smem[];
  float*  S   = (float*)smem;                                // [QB][KW] fp32
  __bf16* Vst = (__bf16*)(smem + QBn * KWn * sizeof(float)); // [DH][KW+8] bf16 transposed
  const int VLD = KWn + 8;

  const int tid  = threadIdx.x;
  const int wave = tid >> 5, lane = tid & 31;
  const int l16  = lane & 15;
  const bool hi  = lane >= 16;

  const int bid  = blockIdx.x;
  const int head = bid >> 6;
  const int rem  = bid & 63;
  const int win  = rem >> 3;
  const int qb   = rem & 7;

  const int    kbase_tok = (win - 1) * WSn;    // negative for win==0 (zero pad)
  const size_t qrow      = (size_t)3 * INNERn;

  // stage V transposed, 4x4 quads: b128 coalesced global loads -> b64 LDS stores
  const float4 fz = {0.f, 0.f, 0.f, 0.f};
  for (int q = tid; q < (DHn / 4) * (KWn / 4); q += 256) {
    int dq = q & 15, jq = q >> 4;
    float4 rv[4];
#pragma unroll
    for (int r = 0; r < 4; ++r) {
      int tk = kbase_tok + 4 * jq + r;
      rv[r] = (tk >= 0)
            ? *(const float4*)&qkv[(size_t)tk * qrow + 2 * INNERn + head * DHn + 4 * dq]
            : fz;
    }
#pragma unroll
    for (int c = 0; c < 4; ++c) {
      v4bf pv;
#pragma unroll
      for (int r = 0; r < 4; ++r) pv[r] = (__bf16)((const float*)&rv[r])[c];
      *(v4bf*)&Vst[(4 * dq + c) * VLD + 4 * jq] = pv;
    }
  }
  __syncthreads();

  // S = (Q K^T) * scale with causal-window mask
  const float scale = 0.125f; // 64^-0.5
  for (int t = wave; t < (QBn / 16) * (KWn / 16); t += 8) {
    int tm = t >> 5;           // 0..1
    int tn = t & 31;           // 0..31
    v8f acc = {};
#pragma unroll
    for (int kd = 0; kd < DHn; kd += 32) {
      int tq = win * WSn + qb * QBn + tm * 16 + l16;
      const float* qp = &qkv[(size_t)tq * qrow + head * DHn + kd + (hi ? 8 : 0)];
      v16bf af = cvt16f(qp, qp + 16);
      v16bf bv;
      int j  = tn * 16 + l16;
      int tk = kbase_tok + j;
      if (tk >= 0) {
        const float* kp = &qkv[(size_t)tk * qrow + INNERn + head * DHn + kd + (hi ? 8 : 0)];
        bv = cvt16f(kp, kp + 16);
      } else {
#pragma unroll
        for (int i = 0; i < 16; ++i) bv[i] = (__bf16)0.f;
      }
      acc = wmma_bf16(af, bv, acc);
    }
#pragma unroll
    for (int r = 0; r < 8; ++r) {
      int i  = tm * 16 + (hi ? 8 : 0) + r;  // 0..31
      int iw = qb * QBn + i;                // row within window 0..255
      int j  = tn * 16 + l16;               // 0..511
      S[i * KWn + j] = (j <= iw + WSn) ? acc[r] * scale : MASKV;
    }
  }
  __syncthreads();

  // row softmax (each wave: 4 rows)
  for (int rr = 0; rr < 4; ++rr) {
    int row = wave * 4 + rr;
    float* Sr = &S[row * KWn];
    float mx = -3.4e38f;
    for (int j = lane; j < KWn; j += 32) mx = fmaxf(mx, Sr[j]);
#pragma unroll
    for (int o = 16; o > 0; o >>= 1) mx = fmaxf(mx, __shfl_xor(mx, o, 32));
    float sum = 0.f;
    for (int j = lane; j < KWn; j += 32) { float e = expf(Sr[j] - mx); Sr[j] = e; sum += e; }
#pragma unroll
    for (int o = 16; o > 0; o >>= 1) sum += __shfl_xor(sum, o, 32);
    float inv = 1.f / sum;
    for (int j = lane; j < KWn; j += 32) Sr[j] *= inv;
  }
  __syncthreads();

  // O = P @ V : 8 output tiles, one per wave
  {
    int tm = wave >> 2;  // 0..1
    int tn = wave & 3;   // 0..3
    v8f acc = {};
    for (int kj = 0; kj < KWn; kj += 32) {
      int i = tm * 16 + l16;
      const float* pp = &S[i * KWn + kj + (hi ? 8 : 0)];
      v16bf af = cvt16f(pp, pp + 16);
      const __bf16* vp = &Vst[(tn * 16 + l16) * VLD + kj + (hi ? 8 : 0)];
      v16bf bv = pack16(*(const v8bf*)vp, *(const v8bf*)(vp + 16));
      acc = wmma_bf16(af, bv, acc);
    }
#pragma unroll
    for (int r = 0; r < 8; ++r) {
      int tq  = win * WSn + qb * QBn + tm * 16 + (hi ? 8 : 0) + r;
      int col = head * DHn + tn * 16 + l16;
      out[(size_t)tq * INNERn + col] = acc[r];
    }
  }
}

// ---------------- layernorm (+ optional token shift of first-half features) ----------------
__global__ __launch_bounds__(256) void ln_kernel(
    const float* __restrict__ in, int in_stride,
    float* __restrict__ out, int out_stride,
    const float* __restrict__ g, const float* __restrict__ b,
    int dim, int nrows, int do_shift)
{
  __shared__ float red[16];
  const int row = blockIdx.x;
  const float* x = in + (size_t)row * in_stride;
  float s = 0.f, ss = 0.f;
  for (int i = threadIdx.x * 4; i < dim; i += 1024) {
    float4 v = *(const float4*)(x + i);
    s  += v.x + v.y + v.z + v.w;
    ss += v.x * v.x + v.y * v.y + v.z * v.z + v.w * v.w;
  }
#pragma unroll
  for (int o = 16; o > 0; o >>= 1) { s += __shfl_xor(s, o, 32); ss += __shfl_xor(ss, o, 32); }
  int wave = threadIdx.x >> 5, lane = threadIdx.x & 31;
  if (lane == 0) { red[wave] = s; red[8 + wave] = ss; }
  __syncthreads();
  if (wave == 0) {
    s  = (lane < 8) ? red[lane] : 0.f;
    ss = (lane < 8) ? red[8 + lane] : 0.f;
#pragma unroll
    for (int o = 4; o > 0; o >>= 1) { s += __shfl_xor(s, o, 32); ss += __shfl_xor(ss, o, 32); }
    if (lane == 0) { red[0] = s; red[1] = ss; }
  }
  __syncthreads();
  float mean = red[0] / dim;
  float var  = red[1] / dim - mean * mean;
  float rstd = rsqrtf(var + 1e-5f);
  int h = dim >> 1;
  for (int i = threadIdx.x; i < dim; i += 256) {
    float v = (x[i] - mean) * rstd * g[i] + b[i];
    if (!do_shift || i >= h) {
      out[(size_t)row * out_stride + i] = v;
    } else {
      if (row + 1 < nrows) out[(size_t)(row + 1) * out_stride + i] = v;
      if (row == 0) out[i] = 0.f;  // shifted-in zeros at row 0
    }
  }
}

// ---------------- small elementwise kernels ----------------
__global__ void embed_kernel(const int* __restrict__ ids, const float* __restrict__ emb,
                             float* __restrict__ x)
{
  int idx = blockIdx.x * blockDim.x + threadIdx.x;
  if (idx >= SEQn * DIMn / 4) return;
  int t = idx >> 8, d4 = (idx & 255) * 4;
  *(float4*)&x[(size_t)t * DIMn + d4] = *(const float4*)&emb[(size_t)ids[t] * DIMn + d4];
}

__global__ void rope_kernel(float* __restrict__ qkv)
{
  int idx = blockIdx.x * blockDim.x + threadIdx.x;
  const int total = SEQn * 3 * HEADSn * (DHn / 2);
  if (idx >= total) return;
  int p = idx & 31;
  int rest = idx >> 5;
  int h = rest & 15;
  rest >>= 4;
  int m = rest % 3;
  int t = rest / 3;
  float inv = expf(-0.28782313662425572f * (float)p);  // 10000^(-2p/64)
  float ang = (float)t * inv;
  float sv = sinf(ang), cv = cosf(ang);
  float2* ptr = (float2*)(qkv + (size_t)t * (3 * INNERn) + m * INNERn + h * DHn + 2 * p);
  float2 v = *ptr;
  float2 o;
  o.x = v.x * cv - v.y * sv;   // x*cos + rot_half(x)*sin
  o.y = v.y * cv + v.x * sv;
  *ptr = o;
}

__global__ void glu_kernel(const float* __restrict__ u, float* __restrict__ a)
{
  int idx = blockIdx.x * blockDim.x + threadIdx.x;
  if (idx >= SEQn * 4096 / 4) return;
  int t = idx >> 10, d4 = (idx & 1023) * 4;
  float4 av = *(const float4*)&u[(size_t)t * 8192 + d4];
  float4 gv = *(const float4*)&u[(size_t)t * 8192 + 4096 + d4];
  float4 o;
  o.x = av.x / (1.f + expf(-gv.x));
  o.y = av.y / (1.f + expf(-gv.y));
  o.z = av.z / (1.f + expf(-gv.z));
  o.w = av.w / (1.f + expf(-gv.w));
  *(float4*)&a[(size_t)t * 4096 + d4] = o;
}

__global__ void sgu_combine_kernel(const float* __restrict__ u, const float* __restrict__ gs,
                                   const float* __restrict__ bias_m, float* __restrict__ out)
{
  int idx = blockIdx.x * blockDim.x + threadIdx.x;
  if (idx >= SEQn * 2048 / 4) return;
  int m = idx >> 9, d4 = (idx & 511) * 4;
  float bm = bias_m[m];
  float4 av = *(const float4*)&u[(size_t)m * 4096 + d4];
  float4 gv = *(const float4*)&gs[(size_t)m * 2048 + d4];
  float4 o;
  o.x = av.x * (gv.x + bm);
  o.y = av.y * (gv.y + bm);
  o.z = av.z * (gv.z + bm);
  o.w = av.w * (gv.w + bm);
  *(float4*)&out[(size_t)m * 2048 + d4] = o;
}

// ---------------- host orchestration ----------------
extern "C" void kernel_launch(void* const* d_in, const int* in_sizes, int n_in,
                              void* d_out, int out_size, void* d_ws, size_t ws_size,
                              hipStream_t stream)
{
  (void)in_sizes; (void)n_in; (void)out_size; (void)ws_size;
  const int*   ids      = (const int*)  d_in[0];
  const float* embed_w  = (const float*)d_in[1];
  const float* attn_ng  = (const float*)d_in[2];
  const float* attn_nb  = (const float*)d_in[3];
  const float* attn_wqkv= (const float*)d_in[4];
  const float* attn_wo  = (const float*)d_in[5];
  const float* attn_bo  = (const float*)d_in[6];
  const float* ffg_ng   = (const float*)d_in[7];
  const float* ffg_nb   = (const float*)d_in[8];
  const float* ffg_win  = (const float*)d_in[9];
  const float* ffg_bin  = (const float*)d_in[10];
  const float* ffg_wout = (const float*)d_in[11];
  const float* ffg_bout = (const float*)d_in[12];
  const float* ffm_ng   = (const float*)d_in[13];
  const float* ffm_nb   = (const float*)d_in[14];
  const float* ffm_win  = (const float*)d_in[15];
  const float* ffm_bin  = (const float*)d_in[16];
  const float* sgu_ng   = (const float*)d_in[17];
  const float* sgu_nb   = (const float*)d_in[18];
  const float* sgu_w    = (const float*)d_in[19];
  const float* sgu_b    = (const float*)d_in[20];
  const float* sgu_pw   = (const float*)d_in[21];
  const float* sgu_pb   = (const float*)d_in[22];
  const float* ffm_wout = (const float*)d_in[23];
  const float* ffm_bout = (const float*)d_in[24];
  const float* fin_ng   = (const float*)d_in[25];
  const float* fin_nb   = (const float*)d_in[26];
  const float* logit_w  = (const float*)d_in[27];
  const float* logit_b  = (const float*)d_in[28];

  float* ws = (float*)d_ws;
  // workspace layout (floats); total 50331648 floats = 192 MB
  float* X   = ws + 0;          // 2048x1024
  float* X2  = ws + 2097152;    // 2048x1024 (residual ping-pong)
  float* H   = ws + 4194304;    // up to 2048x2048 (LN outputs)
  float* QKV = ws + 8388608;    // 2048x3072
  float* O   = ws + 14680064;   // 2048x1024
  float* U   = ws + 16777216;   // up to 2048x8192
  float* AA  = ws + 33554432;   // 2048x4096
  float* T1  = ws + 41943040;   // 2048x2048
  float* T2  = ws + 46137344;   // 2048x2048

  auto gemm = [&](const float* A, const float* B, const float* bias,
                  const float* resid, float* C, int M, int N, int K, int flags) {
    dim3 grid(N / BNt, M / BMt);
    switch (flags) {
      case 0:
        gemm_bf16_kernel<0><<<grid, 256, 0, stream>>>(A, B, bias, resid, C, M, N, K); break;
      case GF_TRIL:
        gemm_bf16_kernel<GF_TRIL><<<grid, 256, 0, stream>>>(A, B, bias, resid, C, M, N, K); break;
      case GF_BIAS:
        gemm_bf16_kernel<GF_BIAS><<<grid, 256, 0, stream>>>(A, B, bias, resid, C, M, N, K); break;
      case GF_BIAS | GF_GELU:
        gemm_bf16_kernel<GF_BIAS | GF_GELU><<<grid, 256, 0, stream>>>(A, B, bias, resid, C, M, N, K); break;
      case GF_BIAS | GF_RESID:
        gemm_bf16_kernel<GF_BIAS | GF_RESID><<<grid, 256, 0, stream>>>(A, B, bias, resid, C, M, N, K); break;
    }
  };
  auto ln = [&](const float* in, int si, float* out, int so,
                const float* g, const float* b, int dim, int shift) {
    ln_kernel<<<SEQn, 256, 0, stream>>>(in, si, out, so, g, b, dim, SEQn, shift);
  };

  // x = embed[ids]
  embed_kernel<<<(SEQn * DIMn / 4) / 256, 256, 0, stream>>>(ids, embed_w, X);

  const size_t attn_smem = (size_t)QBn * KWn * sizeof(float)
                         + (size_t)DHn * (KWn + 8) * sizeof(__bf16);

  for (int i = 0; i < 6; ++i) {
    // ---- windowed attention with rotary ----
    ln(X, DIMn, H, DIMn, attn_ng + i * DIMn, attn_nb + i * DIMn, DIMn, 1);
    gemm(H, attn_wqkv + (size_t)i * DIMn * 3 * INNERn, nullptr, nullptr,
         QKV, SEQn, 3 * INNERn, DIMn, 0);
    rope_kernel<<<(SEQn * 3 * HEADSn * (DHn / 2)) / 256, 256, 0, stream>>>(QKV);
    attn_kernel<<<HEADSn * (SEQn / WSn) * (WSn / QBn), 256, attn_smem, stream>>>(QKV, O);
    gemm(O, attn_wo + (size_t)i * INNERn * DIMn, attn_bo + i * DIMn, X,
         X2, SEQn, DIMn, INNERn, GF_BIAS | GF_RESID);
    { float* t = X; X = X2; X2 = t; }

    if (i < 4) {
      // ---- GLU feedforward ----
      ln(X, DIMn, H, DIMn, ffg_ng + i * DIMn, ffg_nb + i * DIMn, DIMn, 1);
      gemm(H, ffg_win + (size_t)i * DIMn * 8192, ffg_bin + (size_t)i * 8192, nullptr,
           U, SEQn, 8192, DIMn, GF_BIAS);
      glu_kernel<<<(SEQn * 4096 / 4) / 256, 256, 0, stream>>>(U, AA);
      gemm(AA, ffg_wout + (size_t)i * 4096 * DIMn, ffg_bout + i * DIMn, X,
           X2, SEQn, DIMn, 4096, GF_BIAS | GF_RESID);
      { float* t = X; X = X2; X2 = t; }
    } else {
      // ---- gMLP with spatial gating unit ----
      int j = i - 4;
      ln(X, DIMn, H, DIMn, ffm_ng + j * DIMn, ffm_nb + j * DIMn, DIMn, 1);
      gemm(H, ffm_win + (size_t)j * DIMn * 4096, ffm_bin + (size_t)j * 4096, nullptr,
           U, SEQn, 4096, DIMn, GF_BIAS | GF_GELU);
      // g = LN(u[:, 2048:])
      ln(U + 2048, 4096, H, 2048, sgu_ng + (size_t)j * 2048, sgu_nb + (size_t)j * 2048, 2048, 0);
      // spatial: T1 = tril(sgu_w) @ g
      gemm(sgu_w + (size_t)j * SEQn * SEQn, H, nullptr, nullptr,
           T1, SEQn, 2048, SEQn, GF_TRIL);
      // T2 = a * (T1 + sgu_b[m])
      sgu_combine_kernel<<<(SEQn * 2048 / 4) / 256, 256, 0, stream>>>(
          U, T1, sgu_b + (size_t)j * SEQn, T2);
      gemm(T2, sgu_pw + (size_t)j * 2048 * 2048, sgu_pb + (size_t)j * 2048, nullptr,
           T1, SEQn, 2048, 2048, GF_BIAS);
      gemm(T1, ffm_wout + (size_t)j * 2048 * DIMn, ffm_bout + j * DIMn, X,
           X2, SEQn, DIMn, 2048, GF_BIAS | GF_RESID);
      { float* t = X; X = X2; X2 = t; }
    }
  }

  // final LN + logits
  ln(X, DIMn, H, DIMn, fin_ng, fin_nb, DIMn, 0);
  gemm(H, logit_w, logit_b, nullptr, (float*)d_out, SEQn, 512, DIMn, GF_BIAS);
}
